// RNNStack_42992622633441
// MI455X (gfx1250) — matrix operand order repaired and checked by reading, not compiled, not run on hardware
//
#include <hip/hip_runtime.h>
#include <hip/hip_bf16.h>
#include <math.h>

#define B_    128
#define S_    1024
#define EMB_  256
#define HID_  512
#define NST_  2
#define DEP_  12
#define HH_   1024      // 2*HID
#define EPS_  1e-12f
#define THR_  512       // threads per persistent block (16 wave32s)

typedef __bf16 bf16t;
typedef bf16t v16bf __attribute__((ext_vector_type(16)));
typedef float v8f   __attribute__((ext_vector_type(8)));

union Frag { uint4 q[2]; v16bf v; };

__device__ __forceinline__ unsigned short f2bf(float f) {
  unsigned int u = __float_as_uint(f);
  u += 0x7FFFu + ((u >> 16) & 1u);          // round-to-nearest-even
  return (unsigned short)(u >> 16);
}
__device__ __forceinline__ float bf2f(unsigned short h) {
  return __uint_as_float(((unsigned int)h) << 16);
}

__device__ __forceinline__ v8f wmma_bf16(v16bf a, v16bf b, v8f c) {
  // D = A(16x32 bf16) * B(32x16 bf16) + C(16x16 f32)
  return __builtin_amdgcn_wmma_f32_16x16x32_bf16(false, a, false, b, (short)0, c, false, false);
}

// A fragment: 16(M) x 32(K) bf16 from a row-major bf16 buffer (pitch in elems).
// CDNA5 16-bit A layout: lanes 0-15 hold K {0..7,16..23}, lanes 16-31 hold {8..15,24..31}.
__device__ __forceinline__ v16bf load_a(const unsigned short* base, int pitch, int k0, int lane) {
  const int row  = lane & 15;
  const int koff = (lane & 16) ? 8 : 0;
  const unsigned short* p = base + row * pitch + k0 + koff;
  Frag f;
  f.q[0] = *(const uint4*)(p);
  f.q[1] = *(const uint4*)(p + 16);
  return f.v;
}

// B fragment from a pre-swizzled weight buffer: each tile is 32 lanes x 16
// contiguous bf16 (1KB); per-lane data = 32 contiguous bytes.
__device__ __forceinline__ v16bf load_b(const unsigned short* swz, int ktiles, int nt, int kt, int lane) {
  const unsigned short* p = swz + (((size_t)(nt * ktiles + kt)) << 9) + (lane << 4);
  Frag f;
  f.q[0] = *(const uint4*)(p);
  f.q[1] = *(const uint4*)(p + 8);
  return f.v;
}

// ---------------- pre-pass kernels ----------------

// inv[b,e] = 1 / max(||xs[b,:,e]||_2, eps)   (normalize over seq dim, faithful to ref)
__global__ void k_inv(const int* __restrict__ ids, const float* __restrict__ emb,
                      float* __restrict__ inv) {
  const int b = blockIdx.x, e = threadIdx.x;
  const int* idr = ids + b * S_;
  float acc = 0.f;
  for (int s = 0; s < S_; ++s) {
    float v = emb[(size_t)idr[s] * EMB_ + e];
    acc += v * v;
  }
  inv[b * EMB_ + e] = 1.f / fmaxf(sqrtf(acc), EPS_);
}

// xs_bf16[s][b][e] = bf16(emb[id[b,s],e] * inv[b,e])
__global__ void k_xs(const int* __restrict__ ids, const float* __restrict__ emb,
                     const float* __restrict__ inv, unsigned short* __restrict__ xs) {
  const int sId = blockIdx.x, b = blockIdx.y, e = threadIdx.x;
  const int id = ids[b * S_ + sId];
  float v = emb[(size_t)id * EMB_ + e] * inv[b * EMB_ + e];
  xs[(((size_t)sId * B_) + b) * EMB_ + e] = f2bf(v);
}

// Swizzle W (given as [N][K] row-major f32, i.e. GEMM is A[MxK] * W^T) into
// bf16 WMMA B-fragment order: tile(nt,kt) -> 512 elems, elem = lane*16 + j,
// maps to B[k = kt*32 + ((lane&16)>>1) + (j&7) + ((j&8)<<1), n = nt*16 + (lane&15)].
__global__ void k_swz(const float* __restrict__ W, unsigned short* __restrict__ o,
                      int K, int N) {
  const int tid = blockIdx.x * blockDim.x + threadIdx.x;
  if (tid >= K * N) return;
  const int ktiles = K >> 5;
  const int frag = tid >> 9, r = tid & 511;
  const int lane = r >> 4, j = r & 15;
  const int kt = frag % ktiles, nt = frag / ktiles;
  const int k = (kt << 5) + ((lane & 16) >> 1) + (j & 7) + ((j & 8) << 1);
  const int n = (nt << 4) + (lane & 15);
  o[tid] = f2bf(W[(size_t)n * K + k]);
}

// ---------------- persistent scan kernel ----------------
// 8 blocks (one 16-row M-tile each), 512 threads = 16 wave32s, 256 VGPRs/wave.
__global__ __launch_bounds__(THR_, 1)
void k_rnn(const unsigned short* __restrict__ xs,     // [S][B][E] bf16
           const unsigned short* __restrict__ wi_swz, // K=768  N=512
           const unsigned short* __restrict__ wh_swz, // K=512  N=512
           const unsigned short* __restrict__ pv_swz, // K=1024 N=512
           const unsigned short* __restrict__ s1_swz, // K=1024 N=512
           const unsigned short* __restrict__ s2_swz, // K=512  N=256
           const float* __restrict__ pv_b,            // [512]
           const float* __restrict__ ops_w,           // [6][1024]
           float* __restrict__ stacks,                // ping-pong [2][B][2][12][256] f32
           float* __restrict__ out)                   // [B][S][E] f32
{
  __shared__ unsigned short hh[2][16 * HH_];  // hh=[h1|h2] bf16, ping-pong (h_prev = hh[prv][:, :512])
  __shared__ unsigned short peek[16 * 512];   // stack tops, bf16
  __shared__ unsigned short tb[16 * 512];     // relu(hh@sem_w1^T), bf16
  __shared__ float          pvb[16 * 512];    // push values, f32
  __shared__ float          ob[16 * 256];     // sem out before normalize
  __shared__ float          lg[16 * 6];       // ops logits
  __shared__ float          om[16 * 6];       // ops softmax

  const int tid  = threadIdx.x;
  const int lane = tid & 31;
  const int wv   = tid >> 5;          // 0..15
  const int b0   = blockIdx.x << 4;   // first batch row of this block

  unsigned short* hhf = &hh[0][0];
  for (int i = tid; i < 2 * 16 * HH_; i += THR_) hhf[i] = 0;
  for (int i = tid; i < 16 * 512; i += THR_) peek[i] = 0;
  __syncthreads();

  const int cdrow = (lane & 16) ? 8 : 0;  // C/D fragment row offset
  const int ccol  = lane & 15;

  for (int s = 0; s < S_; ++s) {
    const int cur = s & 1, prv = cur ^ 1;
    unsigned short* hhc = &hh[cur][0];
    const unsigned short* hhp = &hh[prv][0];

    // ---- Stage A: ri = [x|peek]@w_i^T, rh = h@w_h^T, hh = [relu(ri+rh)|relu(ri-rh)]
    for (int t2 = 0; t2 < 2; ++t2) {
      const int nt = wv + (t2 << 4);                   // 16 waves x 2 tiles = 512 cols
      v8f ri = {0,0,0,0,0,0,0,0};
      const unsigned short* xrow = xs + ((size_t)s * B_ + b0) * EMB_;
      #pragma unroll 4
      for (int kt = 0; kt < 8; ++kt) {                 // x part: K 0..255
        v16bf a = load_a(xrow, EMB_, kt * 32, lane);
        v16bf b = load_b(wi_swz, 24, nt, kt, lane);
        ri = wmma_bf16(a, b, ri);
      }
      #pragma unroll 4
      for (int kt = 0; kt < 16; ++kt) {                // peek part: K 256..767
        v16bf a = load_a(peek, 512, kt * 32, lane);
        v16bf b = load_b(wi_swz, 24, nt, 8 + kt, lane);
        ri = wmma_bf16(a, b, ri);
      }
      v8f rh = {0,0,0,0,0,0,0,0};
      #pragma unroll 4
      for (int kt = 0; kt < 16; ++kt) {
        v16bf a = load_a(hhp, HH_, kt * 32, lane);     // h_prev = hh_prev[:, :512]
        v16bf b = load_b(wh_swz, 16, nt, kt, lane);
        rh = wmma_bf16(a, b, rh);
      }
      const int col = (nt << 4) + ccol;
      #pragma unroll
      for (int r = 0; r < 8; ++r) {
        const int row = r + cdrow;
        const float x1 = ri[r] + rh[r], x2 = ri[r] - rh[r];
        hhc[row * HH_ + col]         = f2bf(fmaxf(x1, 0.f));
        hhc[row * HH_ + HID_ + col]  = f2bf(fmaxf(x2, 0.f));
      }
    }
    __syncthreads();

    // ---- Stage B: pv = hh@pv_w^T + b (tiles 0..31), t = relu(hh@sem_w1^T) (tiles 32..63)
    for (int t4 = 0; t4 < 4; ++t4) {
      const int tile = wv + (t4 << 4);
      const bool isPv = tile < 32;
      const int nt = isPv ? tile : tile - 32;
      const unsigned short* bw = isPv ? pv_swz : s1_swz;
      v8f acc = {0,0,0,0,0,0,0,0};
      #pragma unroll 4
      for (int kt = 0; kt < 32; ++kt) {
        v16bf a = load_a(hhc, HH_, kt * 32, lane);
        v16bf b = load_b(bw, 32, nt, kt, lane);
        acc = wmma_bf16(a, b, acc);
      }
      const int col = (nt << 4) + ccol;
      if (isPv) {
        const float bias = pv_b[col];
        #pragma unroll
        for (int r = 0; r < 8; ++r) pvb[(r + cdrow) * 512 + col] = acc[r] + bias;
      } else {
        #pragma unroll
        for (int r = 0; r < 8; ++r) tb[(r + cdrow) * 512 + col] = f2bf(fmaxf(acc[r], 0.f));
      }
    }
    // ops logits: 96 length-1024 dots (rows x 2 stacks x 3 ops), 8-wide vectorized
    if (tid < 96) {
      const int row = tid / 6, q = tid % 6;
      const float* wq = ops_w + q * HH_;
      const unsigned short* hr = hhc + row * HH_;
      float acc = 0.f;
      #pragma unroll 4
      for (int h = 0; h < HH_; h += 8) {
        const uint4 hv = *(const uint4*)(hr + h);
        const float4 w0 = *(const float4*)(wq + h);
        const float4 w1 = *(const float4*)(wq + h + 4);
        acc += bf2f((unsigned short)hv.x) * w0.x + bf2f((unsigned short)(hv.x >> 16)) * w0.y
             + bf2f((unsigned short)hv.y) * w0.z + bf2f((unsigned short)(hv.y >> 16)) * w0.w
             + bf2f((unsigned short)hv.z) * w1.x + bf2f((unsigned short)(hv.z >> 16)) * w1.y
             + bf2f((unsigned short)hv.w) * w1.z + bf2f((unsigned short)(hv.w >> 16)) * w1.w;
      }
      lg[row * 6 + q] = acc;
    }
    __syncthreads();

    // softmax over {push,pop,nop}
    if (tid < 32) {
      const int row = tid >> 1, ks = tid & 1;
      const float* l = lg + row * 6 + ks * 3;
      const float m = fmaxf(l[0], fmaxf(l[1], l[2]));
      const float e0 = __expf(l[0] - m), e1 = __expf(l[1] - m), e2 = __expf(l[2] - m);
      const float iv = 1.f / (e0 + e1 + e2);
      float* o = om + row * 6 + ks * 3;
      o[0] = e0 * iv; o[1] = e1 * iv; o[2] = e2 * iv;
    }
    __syncthreads();

    // ---- Stage C: sem out GEMM (1 tile per wave) + stack blend (all threads)
    {
      const int nt = wv;
      v8f acc = {0,0,0,0,0,0,0,0};
      #pragma unroll 4
      for (int kt = 0; kt < 16; ++kt) {
        v16bf a = load_a(tb, 512, kt * 32, lane);
        v16bf b = load_b(s2_swz, 16, nt, kt, lane);
        acc = wmma_bf16(a, b, acc);
      }
      const int col = (nt << 4) + ccol;
      #pragma unroll
      for (int r = 0; r < 8; ++r) ob[(r + cdrow) * 256 + col] = acc[r];
    }
    {
      const int rb = s & 1, wb = rb ^ 1;
      const size_t STK = (size_t)B_ * NST_ * DEP_ * 256;
      for (int i = tid; i < 16 * NST_ * DEP_ * 256; i += THR_) {
        const int e  = i & 255;
        const int d  = (i >> 8) % DEP_;
        const int rk = (i >> 8) / DEP_;        // r*2 + ks
        const int ks = rk & 1, r = rk >> 1;
        const float pu = om[r * 6 + ks * 3 + 0];
        const float po = om[r * 6 + ks * 3 + 1];
        const float nn = om[r * 6 + ks * 3 + 2];
        const size_t base = ((size_t)((b0 + r) * NST_ + ks)) * DEP_ * 256;
        const float* ob_ = stacks + (size_t)rb * STK + base;
        float*       nb_ = stacks + (size_t)wb * STK + base;
        const bool fresh = (s == 0);
        const float od  = fresh ? 0.f : ob_[d * 256 + e];
        const float odm = (d == 0) ? pvb[r * 512 + ks * 256 + e]
                                   : (fresh ? 0.f : ob_[(d - 1) * 256 + e]);
        const float odp = (d == DEP_ - 1) ? 0.f
                                          : (fresh ? 0.f : ob_[(d + 1) * 256 + e]);
        const float nv = pu * odm + po * odp + nn * od;
        nb_[d * 256 + e] = nv;
        if (d == 0) peek[r * 512 + ks * 256 + e] = f2bf(nv);   // next step's A input
      }
    }
    __syncthreads();

    // ---- row-normalize sem out and store to d_out (wave w -> row w)
    {
      const int row = wv;
      const float* orow = ob + row * 256;
      float v[8];
      float s2 = 0.f;
      #pragma unroll
      for (int j = 0; j < 8; ++j) { v[j] = orow[lane * 8 + j]; s2 += v[j] * v[j]; }
      #pragma unroll
      for (int off = 16; off; off >>= 1) s2 += __shfl_xor(s2, off);
      const float sc = 1.f / fmaxf(sqrtf(s2), EPS_);
      float* dst = out + (((size_t)(b0 + row) * S_) + s) * EMB_ + lane * 8;
      *(float4*)(dst)     = make_float4(v[0] * sc, v[1] * sc, v[2] * sc, v[3] * sc);
      *(float4*)(dst + 4) = make_float4(v[4] * sc, v[5] * sc, v[6] * sc, v[7] * sc);
    }
    __syncthreads();
  }
}

extern "C" void kernel_launch(void* const* d_in, const int* in_sizes, int n_in,
                              void* d_out, int out_size, void* d_ws, size_t ws_size,
                              hipStream_t stream) {
  const int*   x_ids  = (const int*)d_in[0];
  const float* emb    = (const float*)d_in[1];
  const float* w_i    = (const float*)d_in[2];
  const float* w_h    = (const float*)d_in[3];
  const float* sem_w1 = (const float*)d_in[4];
  const float* sem_w2 = (const float*)d_in[5];
  const float* pv_w   = (const float*)d_in[6];
  const float* pv_b   = (const float*)d_in[7];
  const float* ops_w  = (const float*)d_in[8];

  char* ws = (char*)d_ws;
  size_t off = 0;
  auto carve = [&](size_t bytes) -> char* {
    char* p = ws + off;
    off += (bytes + 255) & ~(size_t)255;
    return p;
  };
  unsigned short* xs  = (unsigned short*)carve((size_t)S_ * B_ * EMB_ * 2);
  unsigned short* wis = (unsigned short*)carve((size_t)768 * 512 * 2);
  unsigned short* whs = (unsigned short*)carve((size_t)512 * 512 * 2);
  unsigned short* pvs = (unsigned short*)carve((size_t)1024 * 512 * 2);
  unsigned short* s1s = (unsigned short*)carve((size_t)1024 * 512 * 2);
  unsigned short* s2s = (unsigned short*)carve((size_t)512 * 256 * 2);
  float*          inv = (float*)carve((size_t)B_ * EMB_ * 4);
  float*          stk = (float*)carve((size_t)2 * B_ * NST_ * DEP_ * 256 * 4);

  k_inv<<<B_, EMB_, 0, stream>>>(x_ids, emb, inv);
  k_xs<<<dim3(S_, B_), EMB_, 0, stream>>>(x_ids, emb, inv, xs);
  k_swz<<<(768 * 512) / 256, 256, 0, stream>>>(w_i,    wis, 768, 512);
  k_swz<<<(512 * 512) / 256, 256, 0, stream>>>(w_h,    whs, 512, 512);
  k_swz<<<(1024 * 512) / 256, 256, 0, stream>>>(sem_w1, s1s, 1024, 512);
  k_swz<<<(512 * 256) / 256, 256, 0, stream>>>(sem_w2, s2s, 512, 256);
  k_swz<<<(1024 * 512) / 256, 256, 0, stream>>>(pv_w,   pvs, 1024, 512);

  k_rnn<<<B_ / 16, THR_, 0, stream>>>(xs, wis, whs, pvs, s1s, s2s,
                                      pv_b, ops_w, stk, (float*)d_out);
}
